// WeightedAttention_77695958385227
// MI455X (gfx1250) — compile-verified, matrix operand order checked by reading, not run on hardware
//
#include <hip/hip_runtime.h>
#include <hip/hip_bf16.h>

// ---------------------------------------------------------------------------
// MI455X (gfx1250) fused WeightedAttention.
// Compute-bound (~55 GFLOP vs ~55 MB HBM -> ~1000 FLOP/B): all large
// contractions run through v_wmma_f32_16x16x32_bf16 (bf16 operands, f32
// accum). wave32. This revision removes WMMA->VALU WAR hazard NOPs by
// keeping A fragments resident in registers (attn kernel) and by
// grouping/double-buffering fragment loads away from the WMMA bursts.
// ---------------------------------------------------------------------------

typedef __attribute__((ext_vector_type(16))) __bf16 bf16x16;
typedef __attribute__((ext_vector_type(8)))  float  f32x8;

#define WMMA_BF16(A, B, C) \
    __builtin_amdgcn_wmma_f32_16x16x32_bf16(false, (A), false, (B), (short)0, (C), false, false)

__device__ __forceinline__ float sigmoidf(float v) {
    return 1.0f / (1.0f + __expf(-v));
}

// Load a 16-element bf16 fragment slice: elements p[0..7] and p[16..23]
// (per-lane layout of 16-bit A/B WMMA operands). Two 16-byte loads.
__device__ __forceinline__ bf16x16 load_frag16(const __bf16* p) {
    union { bf16x16 v; uint4 u[2]; } t;
    t.u[0] = *(const uint4*)(p);
    t.u[1] = *(const uint4*)(p + 16);
    return t.v;
}

// Same fragment slice but sourced from f32 memory, converted to bf16 in-reg.
__device__ __forceinline__ bf16x16 load_frag_f32(const float* p) {
    bf16x16 f;
#pragma unroll
    for (int j = 0; j < 8; ++j) {
        f[j]     = (__bf16)p[j];
        f[j + 8] = (__bf16)p[j + 16];
    }
    return f;
}

// Sizes
#define B_   2048
#define S_   50
#define D_   128
#define H_   8
#define L_   128
#define HL_  1024   // H*L
#define SD_  6400   // S*D (row stride of x per batch)

// ---------------------------------------------------------------------------
// Kernel 1: weight prep. WT[h][l][d] = bf16(W[h][d][l]); OT[d][j] = bf16(O[j][d]).
// Tiny (2 MB read), runs once per launch. 131072 elements each.
// ---------------------------------------------------------------------------
__global__ __launch_bounds__(256) void wa_prep_kernel(
    const float* __restrict__ Wq, const float* __restrict__ Wk,
    const float* __restrict__ Wv, const float* __restrict__ O,
    __bf16* __restrict__ WqT, __bf16* __restrict__ WkT,
    __bf16* __restrict__ WvT, __bf16* __restrict__ OT) {
    int i = blockIdx.x * 256 + threadIdx.x;
    if (i >= H_ * D_ * L_) return;
    int h = i >> 14, rem = i & 16383, l = rem >> 7, d = rem & 127;
    int src = (h << 14) + (d << 7) + l;     // W[h][d][l]
    WqT[i] = (__bf16)Wq[src];
    WkT[i] = (__bf16)Wk[src];
    WvT[i] = (__bf16)Wv[src];
    int dd = i >> 10, j = i & 1023;         // OT[d][j] = O[j][d]
    OT[i] = (__bf16)O[j * D_ + dd];
}

// ---------------------------------------------------------------------------
// Kernel 2: k = sigmoid(origin @ Wk), GEMM 2048 x 1024 x 128.
// Grid: 256 blocks (32 M-blocks of 64 rows x 8 N-blocks of 128 cols).
// A fragments for a k-step are preloaded as a group before the WMMA burst so
// the 4 back-to-back WMMAs (independent accumulators) absorb the hazard
// window instead of v_nops.
// ---------------------------------------------------------------------------
__global__ __launch_bounds__(256) void wa_kgate_kernel(
    const float* __restrict__ x, const __bf16* __restrict__ WkT,
    float* __restrict__ kbuf) {
    const int mb   = blockIdx.x >> 3;        // 0..31
    const int nb   = blockIdx.x & 7;         // 0..7
    const int wave = threadIdx.x >> 5;
    const int lane = threadIdx.x & 31;
    const int frow = lane & 15;              // fragment row/col within lane
    const int fkb  = (lane >> 4) << 3;       // k sub-offset: 0 or 8

    const int nbase = nb * 128 + wave * 16;
    const __bf16* bCol = WkT + ((nbase + frow) << 7) + fkb;   // stride 128 elems

    f32x8 acc[4] = {};
#pragma unroll
    for (int ks = 0; ks < 4; ++ks) {
        bf16x16 bk = load_frag16(bCol + ks * 32);
        bf16x16 a[4];
#pragma unroll
        for (int mt = 0; mt < 4; ++mt) {
            const float* ap = x + (size_t)(mb * 64 + mt * 16 + frow) * SD_ + ks * 32 + fkb;
            a[mt] = load_frag_f32(ap);       // origin row (s=0), f32 -> bf16
        }
#pragma unroll
        for (int mt = 0; mt < 4; ++mt)
            acc[mt] = WMMA_BF16(a[mt], bk, acc[mt]);
    }
#pragma unroll
    for (int mt = 0; mt < 4; ++mt)
#pragma unroll
        for (int r = 0; r < 8; ++r) {
            int M = mb * 64 + mt * 16 + r + ((lane >> 4) << 3);
            int N = nbase + (lane & 15);
            kbuf[M * HL_ + N] = sigmoidf(acc[mt][r]);
        }
}

// ---------------------------------------------------------------------------
// Kernel 3: fused per-batch attention. One workgroup (8 waves) per b.
// The x-tile A fragments are loaded from LDS ONCE and kept in registers for
// all 8 heads (16 frags = 128 VGPRs): no LDS re-reads, no WAR hazards in the
// WMMA loop. Per head: dual-accumulator WMMA (q & v share A), sigmoid, then
// score -> softmax(S) -> res. LDS: xs 16K + qs 16K + vs 16K + ~1K = ~49 KB.
// ---------------------------------------------------------------------------
__global__ __launch_bounds__(256) void wa_attn_kernel(
    const float* __restrict__ x, const __bf16* __restrict__ WqT,
    const __bf16* __restrict__ WvT, const float* __restrict__ Ws,
    const float* __restrict__ kbuf, __bf16* __restrict__ resB) {
    __shared__ __align__(16) __bf16 xs[64][D_];   // x tile, rows >= S zeroed
    __shared__ __align__(16) __bf16 qs[64][L_];
    __shared__ __align__(16) __bf16 vs[64][L_];
    __shared__ float kw[L_];                      // k[b,h,:] * Ws[h,:]
    __shared__ float sc[64];                      // scores / attn weights
    __shared__ float red[2];                      // max, 1/sum

    const int b    = blockIdx.x;
    const int tid  = threadIdx.x;
    const int wave = tid >> 5;
    const int lane = tid & 31;
    const int frow = lane & 15;
    const int fkb  = (lane >> 4) << 3;

    // Stage x[b] into LDS as bf16, zero-pad rows 50..63 so WMMA K/M padding is exact.
    for (int i = tid; i < 64 * D_; i += 256) {
        int row = i >> 7, col = i & 127;
        xs[row][col] = (row < S_) ? (__bf16)x[(size_t)b * SD_ + row * D_ + col]
                                  : (__bf16)0.0f;
    }
    __syncthreads();

    // Hoist all A fragments (head-invariant) into registers: [ks][mt].
    bf16x16 aF[4][4];
#pragma unroll
    for (int ks = 0; ks < 4; ++ks)
#pragma unroll
        for (int mt = 0; mt < 4; ++mt)
            aF[ks][mt] = load_frag16(&xs[mt * 16 + frow][ks * 32 + fkb]);

    for (int h = 0; h < H_; ++h) {
        if (tid < L_)
            kw[tid] = kbuf[b * HL_ + h * L_ + tid] * Ws[h * L_ + tid];

        // ---- dual GEMM: q and v share A fragments -------------------------
        const __bf16* bqCol = WqT + ((h * L_ + wave * 16 + frow) << 7) + fkb;
        const __bf16* bvCol = WvT + ((h * L_ + wave * 16 + frow) << 7) + fkb;
        f32x8 accq[4] = {}, accv[4] = {};
#pragma unroll
        for (int ks = 0; ks < 4; ++ks) {
            bf16x16 bq = load_frag16(bqCol + ks * 32);
            bf16x16 bv = load_frag16(bvCol + ks * 32);
#pragma unroll
            for (int mt = 0; mt < 4; ++mt) {
                accq[mt] = WMMA_BF16(aF[ks][mt], bq, accq[mt]);
                accv[mt] = WMMA_BF16(aF[ks][mt], bv, accv[mt]);
            }
        }
        // Sigmoid + park in LDS (bf16). C layout: reg r -> M = 16*mt + r + 8*(lane>>4).
#pragma unroll
        for (int mt = 0; mt < 4; ++mt)
#pragma unroll
            for (int r = 0; r < 8; ++r) {
                int srow = mt * 16 + r + ((lane >> 4) << 3);
                int scol = wave * 16 + (lane & 15);
                qs[srow][scol] = (__bf16)sigmoidf(accq[mt][r]);
                vs[srow][scol] = (__bf16)sigmoidf(accv[mt][r]);
            }
        __syncthreads();

        // ---- score[s] = sum_l q[s,l] * k[l] * Ws[l] -----------------------
        if (tid < S_) {
            float s = 0.0f;
#pragma unroll 4
            for (int l = 0; l < L_; ++l) s += (float)qs[tid][l] * kw[l];
            sc[tid] = s;
        }
        __syncthreads();
        if (tid == 0) {                       // softmax over S (50 values: cheap)
            float m = sc[0];
            for (int s = 1; s < S_; ++s) m = fmaxf(m, sc[s]);
            red[0] = m;
        }
        __syncthreads();
        if (tid < S_) sc[tid] = __expf(sc[tid] - red[0]);
        __syncthreads();
        if (tid == 0) {
            float sum = 0.0f;
            for (int s = 0; s < S_; ++s) sum += sc[s];
            red[1] = 1.0f / sum;
        }
        __syncthreads();

        // ---- res[l] = (1/sum) * sum_s attn[s] * v[s,l] --------------------
        if (tid < L_) {
            float r = 0.0f;
#pragma unroll 2
            for (int s = 0; s < S_; ++s) r += sc[s] * (float)vs[s][tid];
            resB[b * HL_ + h * L_ + tid] = (__bf16)(r * red[1]);
        }
        __syncthreads();   // qs/vs/kw/sc reused next head
    }
}

// ---------------------------------------------------------------------------
// Kernel 4: out = sigmoid(res @ O) + origin. GEMM 2048 x 128 x 1024.
// Grid 32 blocks of 64 rows; 8 waves cover the 128 output cols (1 N-tile
// each). Fragments are double-buffered: loads for k-step ks+1 issue before
// the WMMA burst of k-step ks, covering load latency and hazard windows.
// ---------------------------------------------------------------------------
__global__ __launch_bounds__(256) void wa_out_kernel(
    const __bf16* __restrict__ resB, const __bf16* __restrict__ OT,
    const float* __restrict__ x, float* __restrict__ out) {
    const int mb   = blockIdx.x;             // 0..31
    const int wave = threadIdx.x >> 5;
    const int lane = threadIdx.x & 31;
    const int frow = lane & 15;
    const int fkb  = (lane >> 4) << 3;

    const __bf16* bCol = OT + ((wave * 16 + frow) << 10) + fkb;   // stride 1024
    const __bf16* aRow[4];
#pragma unroll
    for (int mt = 0; mt < 4; ++mt)
        aRow[mt] = resB + ((mb * 64 + mt * 16 + frow) << 10) + fkb;

    bf16x16 aBuf[2][4], bBuf[2];
    bBuf[0] = load_frag16(bCol);
#pragma unroll
    for (int mt = 0; mt < 4; ++mt) aBuf[0][mt] = load_frag16(aRow[mt]);

    f32x8 acc[4] = {};
    for (int ks = 0; ks < 32; ++ks) {
        const int cur = ks & 1, nxt = cur ^ 1;
        if (ks < 31) {
            bBuf[nxt] = load_frag16(bCol + (ks + 1) * 32);
#pragma unroll
            for (int mt = 0; mt < 4; ++mt)
                aBuf[nxt][mt] = load_frag16(aRow[mt] + (ks + 1) * 32);
        }
#pragma unroll
        for (int mt = 0; mt < 4; ++mt)
            acc[mt] = WMMA_BF16(aBuf[cur][mt], bBuf[cur], acc[mt]);
    }
#pragma unroll
    for (int mt = 0; mt < 4; ++mt)
#pragma unroll
        for (int r = 0; r < 8; ++r) {
            int M = mb * 64 + mt * 16 + r + ((lane >> 4) << 3);
            int d = wave * 16 + (lane & 15);
            out[M * D_ + d] = sigmoidf(acc[mt][r]) + x[(size_t)M * SD_ + d];
        }
}

// ---------------------------------------------------------------------------
// Launch. Workspace layout (bytes):
//   0        WqT bf16 (256K)     262144  WkT bf16 (256K)
//   524288   WvT bf16 (256K)     786432  OT  bf16 (256K)
//   1048576  kbuf f32 (8M)       9437184 resB bf16 (4M)     total ~13.5 MB
// ---------------------------------------------------------------------------
extern "C" void kernel_launch(void* const* d_in, const int* in_sizes, int n_in,
                              void* d_out, int out_size, void* d_ws, size_t ws_size,
                              hipStream_t stream) {
    const float* x  = (const float*)d_in[0];
    const float* Wq = (const float*)d_in[1];
    const float* Wk = (const float*)d_in[2];
    const float* Wv = (const float*)d_in[3];
    const float* Ws = (const float*)d_in[4];
    const float* O  = (const float*)d_in[5];
    float* out = (float*)d_out;

    char* ws = (char*)d_ws;
    __bf16* WqT  = (__bf16*)(ws);
    __bf16* WkT  = (__bf16*)(ws + 262144);
    __bf16* WvT  = (__bf16*)(ws + 524288);
    __bf16* OT   = (__bf16*)(ws + 786432);
    float*  kbuf = (float*)(ws + 1048576);
    __bf16* resB = (__bf16*)(ws + 9437184);

    wa_prep_kernel<<<512, 256, 0, stream>>>(Wq, Wk, Wv, O, WqT, WkT, WvT, OT);
    wa_kgate_kernel<<<256, 256, 0, stream>>>(x, WkT, kbuf);
    wa_attn_kernel<<<B_, 256, 0, stream>>>(x, WqT, WvT, Ws, kbuf, resB);
    wa_out_kernel<<<32, 256, 0, stream>>>(resB, OT, x, out);
}
